// LSTMSeq_29205777613319
// MI455X (gfx1250) — matrix operand order repaired and checked by reading, not compiled, not run on hardware
//
#include <hip/hip_runtime.h>
#include <cstdint>
#include <cstddef>

// ---------------------------------------------------------------------------
// Types
// ---------------------------------------------------------------------------
typedef __attribute__((ext_vector_type(8)))  unsigned short v8us;
typedef __attribute__((ext_vector_type(8)))  __bf16         v8bf;
typedef __attribute__((ext_vector_type(16))) __bf16         v16bf;
typedef __attribute__((ext_vector_type(8)))  float          v8f;
typedef __attribute__((ext_vector_type(4)))  int            v4i;

static constexpr int kBatch = 8;
static constexpr int kDim   = 256;
static constexpr int kS     = 4096;          // 64*64
static constexpr int kHid   = 512;
static constexpr int kGate  = 2048;          // 4*kHid
static constexpr int kM     = kBatch * kS;   // 32768 rows for the dense GEMMs
static constexpr int kNWG   = 16;            // persistent WGs in scan

// ---------------------------------------------------------------------------
// Async global->LDS copy (CDNA5 GLOBAL_LOAD_ASYNC_TO_LDS_B128, ASYNCcnt)
// Builtin signature (from hipcc diagnostic): (v4i AS1*, v4i AS3*, imm, imm)
// ---------------------------------------------------------------------------
#if defined(__has_builtin)
#if __has_builtin(__builtin_amdgcn_global_load_async_to_lds_b128)
#define USE_ASYNC_LDS 1
#endif
#endif

typedef __attribute__((address_space(1))) v4i as1_v4i;
typedef __attribute__((address_space(3))) v4i as3_v4i;

__device__ __forceinline__ void g2l_b128(const void* g, void* l) {
#ifdef USE_ASYNC_LDS
  __builtin_amdgcn_global_load_async_to_lds_b128((as1_v4i*)g, (as3_v4i*)l, 0, 0);
#else
  *(v8us*)l = *(const v8us*)g;
#endif
}

__device__ __forceinline__ void g2l_wait() {
#ifdef USE_ASYNC_LDS
#if __has_builtin(__builtin_amdgcn_s_wait_asynccnt)
  __builtin_amdgcn_s_wait_asynccnt(0);
#else
  asm volatile("s_wait_asynccnt 0" ::: "memory");
#endif
#endif
}

// ---------------------------------------------------------------------------
// Helpers
// ---------------------------------------------------------------------------
__device__ __forceinline__ unsigned short f2bf(float f) {
  unsigned u = __builtin_bit_cast(unsigned, f);
  u += 0x7FFFu + ((u >> 16) & 1u);       // round-to-nearest-even
  return (unsigned short)(u >> 16);
}

// CDNA5 native V_TANH_F32 (single TRANS op) -- keeps the serial scan path
// short instead of the branchy libm range-reduction expansion.
__device__ __forceinline__ float fast_tanh(float x) {
#if defined(__has_builtin)
#if __has_builtin(__builtin_amdgcn_tanhf)
  return __builtin_amdgcn_tanhf(x);
#elif __has_builtin(__builtin_amdgcn_tanh_f32)
  return __builtin_amdgcn_tanh_f32(x);
#else
  return tanhf(x);
#endif
#else
  return tanhf(x);
#endif
}

// sigmoid(x) = 0.5 + 0.5 * tanh(0.5 x): 1 mul + 1 v_tanh + 1 fma.
__device__ __forceinline__ float sigm(float x) {
  return 0.5f + 0.5f * fast_tanh(0.5f * x);
}

// A-matrix fragment (16x32 bf16, M x K): lane<16 holds row m, K = k0..k0+7 and
// k0+16..23 (VGPR0-3 / 4-7); lane>=16 holds K = k0+8..15 and k0+24..31.
template <typename P>
__device__ __forceinline__ v16bf frag_a(P base, int ld, int lane) {
  const int m = lane & 15, sel = lane >> 4;
  P p = base + (size_t)m * ld + sel * 8;
  v8bf lo = *(const v8bf*)(p);
  v8bf hi = *(const v8bf*)(p + 16);
  return __builtin_shufflevector(lo, hi, 0,1,2,3,4,5,6,7,8,9,10,11,12,13,14,15);
}

// B-matrix fragment (32x16 bf16, K x N). Our data is W[N][K] row-major and
// B[k][n] = W[n][k]: lane<16 -> column n = lane, K = k0..k0+7 and k0+16..23;
// lane>=16 -> K = k0+8..15 and k0+24..31 (same K order as frag_a).
template <typename P>
__device__ __forceinline__ v16bf frag_b(P base, int ld, int lane) {
  const int n = lane & 15, sel = lane >> 4;
  P p = base + (size_t)n * ld + sel * 8;
  v8bf lo = *(const v8bf*)(p);
  v8bf hi = *(const v8bf*)(p + 16);
  return __builtin_shufflevector(lo, hi, 0,1,2,3,4,5,6,7,8,9,10,11,12,13,14,15);
}

__device__ __forceinline__ v8f wmma_bf16(v16bf a, v16bf b, v8f c) {
  return __builtin_amdgcn_wmma_f32_16x16x32_bf16(false, a, false, b, (short)0, c,
                                                 false, false);
}

__device__ __forceinline__ void grid_barrier(unsigned* bar, unsigned target) {
  __syncthreads();
  if (threadIdx.x == 0) {
    __threadfence();
    __hip_atomic_fetch_add(bar, 1u, __ATOMIC_RELEASE, __HIP_MEMORY_SCOPE_AGENT);
    while (__hip_atomic_load(bar, __ATOMIC_ACQUIRE, __HIP_MEMORY_SCOPE_AGENT) < target)
      __builtin_amdgcn_s_sleep(2);
  }
  __syncthreads();
}

// ---------------------------------------------------------------------------
// Prep kernels
// ---------------------------------------------------------------------------
__global__ void init_bar_kernel(unsigned* bar) {
  if (threadIdx.x < 128) bar[threadIdx.x] = 0u;
}

__global__ void cvt_bf16_kernel(const float* __restrict__ in,
                                unsigned short* __restrict__ out, int n) {
  for (int i = blockIdx.x * blockDim.x + threadIdx.x; i < n; i += gridDim.x * blockDim.x)
    out[i] = f2bf(in[i]);
}

// x [B, C, H*W] fp32  ->  xseq [B, H*W, C] bf16
__global__ void pack_xseq_kernel(const float* __restrict__ x,
                                 unsigned short* __restrict__ xseq) {
  int idx = blockIdx.x * blockDim.x + threadIdx.x;   // b*S*C + s*C + c
  int c = idx & (kDim - 1);
  int s = (idx >> 8) & (kS - 1);
  int b = idx >> 20;
  xseq[idx] = f2bf(x[((size_t)(b * kDim + c) << 12) + s]);
}

// ---------------------------------------------------------------------------
// Dense TN GEMM: C[M,N] = A[M,K] (bf16) * W[N,K]^T (bf16) + bias
// OUT_MODE 0: fp32 store          (gate pre-activations gx)
// OUT_MODE 1: relu -> bf16 store  (fc1)
// OUT_MODE 2: fp32 store transposed to [B, N, S] layout (fc2 -> final output)
// Block: 256 thr = 8 waves, tile 32(M) x 256(N); wave tile 16 x 64.
// ---------------------------------------------------------------------------
template <int OUT_MODE>
__global__ __launch_bounds__(256)
void gemm_tn_kernel(const unsigned short* __restrict__ A,
                    const unsigned short* __restrict__ W,
                    const float* __restrict__ bias,
                    void* __restrict__ Cout, int N, int K) {
  const int tid  = threadIdx.x;
  const int lane = tid & 31;
  const int wv   = tid >> 5;
  const int m0   = blockIdx.x * 32 + (wv & 1) * 16;
  const int n0   = blockIdx.y * 256 + (wv >> 1) * 64;

  v8f acc[4];
#pragma unroll
  for (int i = 0; i < 4; ++i)
#pragma unroll
    for (int r = 0; r < 8; ++r) acc[i][r] = 0.0f;

#pragma unroll 2
  for (int kb = 0; kb < K; kb += 32) {
    // Pull the next A K-panel toward the caches while this one computes
    // (lowers to global_prefetch_b8).
    if (kb + 64 < K)
      __builtin_prefetch(A + (size_t)(m0 + (lane & 15)) * K + kb + 64, 0, 0);
    v16bf af = frag_a(A + (size_t)m0 * K + kb, K, lane);
#pragma unroll
    for (int i = 0; i < 4; ++i) {
      v16bf bf = frag_b(W + (size_t)(n0 + 16 * i) * K + kb, K, lane);
      acc[i] = wmma_bf16(af, bf, acc[i]);
    }
  }

  const int n = lane & 15, sel = lane >> 4;
#pragma unroll
  for (int i = 0; i < 4; ++i) {
    const int col = n0 + 16 * i + n;
    const float bv = bias[col];
#pragma unroll
    for (int r = 0; r < 8; ++r) {
      const int row = m0 + r + 8 * sel;
      float v = acc[i][r] + bv;
      if (OUT_MODE == 0) {
        ((float*)Cout)[(size_t)row * N + col] = v;
      } else if (OUT_MODE == 1) {
        v = fmaxf(v, 0.0f);
        ((unsigned short*)Cout)[(size_t)row * N + col] = f2bf(v);
      } else {
        const int b = row >> 12, s = row & (kS - 1);
        ((float*)Cout)[(((size_t)b << 8) + col) * kS + s] = v;
      }
    }
  }
}

// ---------------------------------------------------------------------------
// Persistent LSTM scan. Grid = 16 WGs x 256 thr (8 waves). WG `wg` owns hidden
// units j in [32*wg, 32*wg+32): gate rows {g*512 + 32*wg + j} for g=i,f,g,o.
// Its 128x512 bf16 W_hh slice lives in LDS for the whole scan (128 KB).
// Per step: async-stage h(t-1) [16x512 bf16, rows 8..15 zero] into LDS,
// 8 waves x 16 K-steps of v_wmma, gate math with register cell state, publish
// h slice to ping-pong global buffer, grid barrier.
// ---------------------------------------------------------------------------
__global__ __launch_bounds__(256)
void lstm_scan_kernel(const unsigned short* __restrict__ Whh,  // [2048,512] bf16
                      const float* __restrict__ gx,            // [B,S,2048] fp32 (bias folded)
                      unsigned short* __restrict__ hseq,       // [B,S,512] bf16 out
                      unsigned short* __restrict__ hbuf,       // [2][16][512] bf16 ping-pong
                      unsigned* __restrict__ bar) {
  extern __shared__ char smem[];
  unsigned short* wlds = (unsigned short*)smem;                 // 128*512
  unsigned short* hlds = wlds + 128 * 512;                      // 16*512
  float* gatebuf = (float*)(hlds + 16 * 512);                   // 16*128

  const int tid  = threadIdx.x;
  const int wg   = blockIdx.x;           // 0..15
  const int lane = tid & 31;
  const int wv   = tid >> 5;             // 0..7

  // Preload W_hh slice via async engine: lds row r -> global gate row
  // (r>>5)*512 + wg*32 + (r&31).  128 KB = 8192 b128 transfers.
  for (int e = tid; e < 128 * 64; e += 256) {
    const int r = e >> 6, ch = e & 63;
    const int grow = ((r >> 5) << 9) + (wg << 5) + (r & 31);
    g2l_b128(&Whh[(size_t)grow * 512 + ch * 8], &wlds[r * 512 + ch * 8]);
  }
  // Zero ping-pong h buffers (covers padded batch rows 8..15 forever).
  for (int e = tid; e < 1024; e += 256) hbuf[wg * 1024 + e] = 0;
  g2l_wait();
  grid_barrier(bar, kNWG);

  float cstate = 0.0f;
  const int eb = tid >> 5;               // batch 0..7
  const int ej = tid & 31;               // hidden unit within slice

  for (int t = 0; t < kS; ++t) {
    // Async-stage h(t-1) into LDS (16 KB, 4 b128 transfers per thread).
    const unsigned short* hprev = hbuf + ((t & 1) ? 16 * 512 : 0);
    for (int e = tid; e < 1024; e += 256)
      g2l_b128(&hprev[e * 8], &hlds[e * 8]);
    g2l_wait();
    __syncthreads();

    // Recurrent matmul: [16 x 512] x [512 x 128-slice]; wave wv -> 16 cols.
    v8f acc;
#pragma unroll
    for (int r = 0; r < 8; ++r) acc[r] = 0.0f;
    const int n0 = wv * 16;
#pragma unroll 4
    for (int kb = 0; kb < kHid; kb += 32) {
      v16bf af = frag_a(hlds + kb, 512, lane);
      v16bf bf = frag_b(wlds + (size_t)n0 * 512 + kb, 512, lane);
      acc = wmma_bf16(af, bf, acc);
    }
    const int n = lane & 15, sel = lane >> 4;
#pragma unroll
    for (int r = 0; r < 8; ++r)
      gatebuf[(r + 8 * sel) * 128 + n0 + n] = acc[r];
    __syncthreads();

    // Gate nonlinearities + cell update; one (batch, hidden) pair per thread.
    {
      const float* gxp = gx + ((size_t)(eb * kS + t) << 11);   // *2048
      const int jj = (wg << 5) + ej;
      float iv = gatebuf[eb * 128 +       ej] + gxp[jj];
      float fv = gatebuf[eb * 128 +  32 + ej] + gxp[512 + jj];
      float gv = gatebuf[eb * 128 +  64 + ej] + gxp[1024 + jj];
      float ov = gatebuf[eb * 128 +  96 + ej] + gxp[1536 + jj];
      iv = sigm(iv); fv = sigm(fv); gv = fast_tanh(gv); ov = sigm(ov);
      cstate = fv * cstate + iv * gv;
      const unsigned short hb = f2bf(ov * fast_tanh(cstate));
      unsigned short* hw = hbuf + ((t & 1) ? 0 : 16 * 512);
      hw[eb * 512 + jj] = hb;
      hseq[((size_t)(eb * kS + t) << 9) + jj] = hb;
    }
    grid_barrier(bar, (unsigned)(kNWG * (t + 2)));
  }
}

// ---------------------------------------------------------------------------
// Host-side orchestration
// ---------------------------------------------------------------------------
extern "C" void kernel_launch(void* const* d_in, const int* in_sizes, int n_in,
                              void* d_out, int out_size, void* d_ws, size_t ws_size,
                              hipStream_t stream) {
  (void)in_sizes; (void)n_in; (void)out_size; (void)ws_size;
  const float* x    = (const float*)d_in[0];
  const float* Wih0 = (const float*)d_in[1];
  const float* Whh0 = (const float*)d_in[2];
  const float* b0   = (const float*)d_in[3];
  const float* Wih1 = (const float*)d_in[4];
  const float* Whh1 = (const float*)d_in[5];
  const float* b1   = (const float*)d_in[6];
  const float* fc1w = (const float*)d_in[7];
  const float* fc1b = (const float*)d_in[8];
  const float* fc2w = (const float*)d_in[9];
  const float* fc2b = (const float*)d_in[10];

  char* ws = (char*)d_ws;
  size_t off = 0;
  auto carve = [&](size_t bytes) -> void* {
    void* p = ws + off;
    off = (off + bytes + 511) & ~(size_t)511;
    return p;
  };
  unsigned*       bar     = (unsigned*)      carve(512);                         // 2 counters
  unsigned short* hbuf    = (unsigned short*)carve(2 * 16 * 512 * 2);            // 32 KB
  unsigned short* xseq    = (unsigned short*)carve((size_t)kM * kDim * 2);       // 16 MB
  unsigned short* Wih0b   = (unsigned short*)carve((size_t)kGate * kDim * 2);    // 1 MB
  unsigned short* Whh0b   = (unsigned short*)carve((size_t)kGate * kHid * 2);    // 2 MB
  unsigned short* Wih1b   = (unsigned short*)carve((size_t)kGate * kHid * 2);    // 2 MB
  unsigned short* Whh1b   = (unsigned short*)carve((size_t)kGate * kHid * 2);    // 2 MB
  unsigned short* fc1b16  = (unsigned short*)carve((size_t)kDim * kHid * 2);     // 256 KB
  unsigned short* fc2b16  = (unsigned short*)carve((size_t)kDim * kDim * 2);     // 128 KB
  float*          gx      = (float*)         carve((size_t)kM * kGate * 4);      // 256 MB (reused)
  unsigned short* h0seq   = (unsigned short*)carve((size_t)kM * kHid * 2);       // 32 MB
  unsigned short* h1seq   = (unsigned short*)carve((size_t)kM * kHid * 2);       // 32 MB
  unsigned short* y1      = (unsigned short*)carve((size_t)kM * kDim * 2);       // 16 MB

  // 0) reset grid-barrier counters (graph-capture safe, deterministic)
  init_bar_kernel<<<1, 128, 0, stream>>>(bar);

  // 1) precision conversion / layout packing
  pack_xseq_kernel<<<kM * kDim / 256, 256, 0, stream>>>(x, xseq);
  cvt_bf16_kernel<<<512, 256, 0, stream>>>(Wih0, Wih0b, kGate * kDim);
  cvt_bf16_kernel<<<512, 256, 0, stream>>>(Whh0, Whh0b, kGate * kHid);
  cvt_bf16_kernel<<<512, 256, 0, stream>>>(Wih1, Wih1b, kGate * kHid);
  cvt_bf16_kernel<<<512, 256, 0, stream>>>(Whh1, Whh1b, kGate * kHid);
  cvt_bf16_kernel<<<128, 256, 0, stream>>>(fc1w, fc1b16, kDim * kHid);
  cvt_bf16_kernel<<<64,  256, 0, stream>>>(fc2w, fc2b16, kDim * kDim);

  const size_t scan_lds = 128 * 512 * 2 + 16 * 512 * 2 + 16 * 128 * 4; // 152 KB

  // 2) gx0 = xseq @ Wih0^T + b0   [32768 x 2048], K=256
  gemm_tn_kernel<0><<<dim3(kM / 32, kGate / 256), 256, 0, stream>>>(
      xseq, Wih0b, b0, gx, kGate, kDim);

  // 3) layer-0 recurrence
  lstm_scan_kernel<<<kNWG, 256, scan_lds, stream>>>(Whh0b, gx, h0seq, hbuf, bar);

  // 4) gx1 = h0seq @ Wih1^T + b1   [32768 x 2048], K=512
  gemm_tn_kernel<0><<<dim3(kM / 32, kGate / 256), 256, 0, stream>>>(
      h0seq, Wih1b, b1, gx, kGate, kHid);

  // 5) layer-1 recurrence
  lstm_scan_kernel<<<kNWG, 256, scan_lds, stream>>>(Whh1b, gx, h1seq, hbuf, bar + 64);

  // 6) y1 = relu(h1seq @ fc1^T + fc1b)  -> bf16 [32768 x 256], K=512
  gemm_tn_kernel<1><<<dim3(kM / 32, 1), 256, 0, stream>>>(
      h1seq, fc1b16, fc1b, y1, kDim, kHid);

  // 7) out = (y1 @ fc2^T + fc2b) transposed to [B, C, H, W] fp32, K=256
  gemm_tn_kernel<2><<<dim3(kM / 32, 1), 256, 0, stream>>>(
      y1, fc2b16, fc2b, d_out, kDim, kDim);
}
